// SimpleRNN_30064771072670
// MI455X (gfx1250) — matrix operand order repaired and checked by reading, not compiled, not run on hardware
//
#include <hip/hip_runtime.h>
#include <hip/hip_bf16.h>

typedef _Float16 v8h  __attribute__((ext_vector_type(8)));
typedef _Float16 v16h __attribute__((ext_vector_type(16)));
typedef float    v8f  __attribute__((ext_vector_type(8)));

constexpr int Bc = 256, Tc = 512, Ic = 64, Hc = 256;
constexpr int KTOT = Hc + Ic;   // 320: [h ; x_t]
constexpr int NKT  = KTOT / 32; // 10 K-tiles of 32
constexpr int ROWH = 328;       // padded LDS row length in f16 (656 B = 41*16 B)
constexpr int MT   = 16;        // batch rows per workgroup

__device__ __forceinline__ v16h cat8(v8h lo, v8h hi) {
    return __builtin_shufflevector(lo, hi, 0,1,2,3,4,5,6,7,8,9,10,11,12,13,14,15);
}

// load 16 consecutive f32 (64B-aligned) and convert to a f16 fragment vector
__device__ __forceinline__ v16h cvt16(const float* p) {
    v16h r;
    #pragma unroll
    for (int q = 0; q < 4; ++q) {
        float4 f = ((const float4*)p)[q];
        r[4*q+0] = (_Float16)f.x; r[4*q+1] = (_Float16)f.y;
        r[4*q+2] = (_Float16)f.z; r[4*q+3] = (_Float16)f.w;
    }
    return r;
}

// tanh(x) = 1 - 2/(2^(x*2*log2(e)) + 1)
// Exact at saturation (exp2->inf => 1, exp2->0 => -1); ~5 VALU via v_exp_f32/v_rcp_f32.
__device__ __forceinline__ float fast_tanh(float xv) {
    float e = __builtin_amdgcn_exp2f(xv * 2.885390081777927f); // 2*log2(e)
    return 1.0f - 2.0f * __builtin_amdgcn_rcpf(e + 1.0f);
}

__global__ void __launch_bounds__(256)
__attribute__((amdgpu_waves_per_eu(1, 4)))
rnn_fused_kernel(const float* __restrict__ x,
                 const float* __restrict__ Wih,
                 const float* __restrict__ Whh,
                 const float* __restrict__ bih,
                 const float* __restrict__ bhh,
                 const float* __restrict__ fcW,
                 const float* __restrict__ fcb,
                 float* __restrict__ out)
{
    // double-buffered activations [h(256) | x_t(64)] for 16 batch rows (~21 KB)
    __shared__ _Float16 act[2][MT * ROWH];

    const int tid  = threadIdx.x;
    const int lane = tid & 31;
    const int wave = tid >> 5;
    const int half = lane >> 4;
    const int l16  = lane & 15;
    const int b0   = blockIdx.x * MT;

    // ---- h0 = 0 ----
    for (int idx = tid; idx < MT * Hc; idx += 256) {
        int m = idx >> 8, k = idx & 255;
        act[0][m * ROWH + k] = (_Float16)0.0f;
    }
    // ---- stage x_0 into act[0][:, 256:320] ----
    {
        int base = tid * 4, row = base >> 6, col = base & 63;
        const float4 v = *(const float4*)(x + ((size_t)(b0 + row) * Tc + 0) * Ic + col);
        _Float16* dst = &act[0][row * ROWH + Hc + col];
        dst[0] = (_Float16)v.x; dst[1] = (_Float16)v.y;
        dst[2] = (_Float16)v.z; dst[3] = (_Float16)v.w;
    }

    // ---- per-wave N-tiles: weight-stationary B fragments in registers ----
    // B frag for ktile kt, half h: 16 contiguous K values starting at K = kt*32 + h*16,
    // for output row n (lane = n%16).  K<256 -> W_hh[n][K], K>=256 -> W_ih[n][K-256].
    const int   n0 = (wave * 2 + 0) * 16 + l16;
    const int   n1 = (wave * 2 + 1) * 16 + l16;
    const float bias0 = bih[n0] + bhh[n0];
    const float bias1 = bih[n1] + bhh[n1];

    v16h wreg0[NKT], wreg1[NKT];
    #pragma unroll
    for (int kt = 0; kt < NKT; ++kt) {
        const int koff = kt * 32 + half * 16;
        const float* p0 = (koff < Hc) ? (Whh + n0 * Hc + koff)
                                      : (Wih + n0 * Ic + (koff - Hc));
        const float* p1 = (koff < Hc) ? (Whh + n1 * Hc + koff)
                                      : (Wih + n1 * Ic + (koff - Hc));
        wreg0[kt] = cvt16(p0);
        wreg1[kt] = cvt16(p1);
    }

    __syncthreads();

    for (int t = 0; t < Tc; ++t) {
        _Float16* cur = act[t & 1];
        _Float16* nxt = act[(t + 1) & 1];

        v8f c0 = {}; v8f c1 = {};
        const _Float16* arow = cur + l16 * ROWH + half * 16; // A frag row m = l16

        #pragma unroll
        for (int kt = 0; kt < NKT; ++kt) {
            v8h alo = *(const v8h*)(arow + kt * 32);        // K = kt*32 + half*16 .. +7
            v8h ahi = *(const v8h*)(arow + kt * 32 + 8);    // K = +8 .. +15
            v16h a  = cat8(alo, ahi);
            c0 = __builtin_amdgcn_wmma_f32_16x16x32_f16(
                     false, a, false, wreg0[kt], (short)0, c0, false, false);
            c1 = __builtin_amdgcn_wmma_f32_16x16x32_f16(
                     false, a, false, wreg1[kt], (short)0, c1, false, false);
        }

        // stage x_{t+1} into the next buffer while accumulators settle
        if (t + 1 < Tc) {
            int base = tid * 4, row = base >> 6, col = base & 63;
            const float4 v = *(const float4*)(x + ((size_t)(b0 + row) * Tc + (t + 1)) * Ic + col);
            _Float16* dst = nxt + row * ROWH + Hc + col;
            dst[0] = (_Float16)v.x; dst[1] = (_Float16)v.y;
            dst[2] = (_Float16)v.z; dst[3] = (_Float16)v.w;
        }

        // h_new = tanh(acc + bias); C/D layout: lane holds (m = r + 8*half, n = l16)
        #pragma unroll
        for (int r = 0; r < 8; ++r) {
            int m = r + half * 8;
            nxt[m * ROWH + n0] = (_Float16)fast_tanh(c0[r] + bias0);
            nxt[m * ROWH + n1] = (_Float16)fast_tanh(c1[r] + bias1);
        }
        __syncthreads();
    }

    // final h lives in act[0] (Tc even)
    const _Float16* fin = act[Tc & 1];
    if (tid < MT) {
        float acc = fcb[0];
        for (int k = 0; k < Hc; ++k)
            acc += (float)fin[tid * ROWH + k] * fcW[k];
        out[b0 + tid] = acc;
    }
}

extern "C" void kernel_launch(void* const* d_in, const int* in_sizes, int n_in,
                              void* d_out, int out_size, void* d_ws, size_t ws_size,
                              hipStream_t stream) {
    (void)in_sizes; (void)n_in; (void)out_size; (void)d_ws; (void)ws_size;
    const float* x   = (const float*)d_in[0];
    const float* Wih = (const float*)d_in[1];
    const float* Whh = (const float*)d_in[2];
    const float* bih = (const float*)d_in[3];
    const float* bhh = (const float*)d_in[4];
    const float* fcW = (const float*)d_in[5];
    const float* fcb = (const float*)d_in[6];
    float* out = (float*)d_out;

    rnn_fused_kernel<<<dim3(Bc / MT), dim3(256), 0, stream>>>(
        x, Wih, Whh, bih, bhh, fcW, fcb, out);
}
